// GATCNN_40888088658032
// MI455X (gfx1250) — matrix operand order repaired and checked by reading, not compiled, not run on hardware
//
#include <hip/hip_runtime.h>

typedef __attribute__((ext_vector_type(16))) __bf16 v16bf;
typedef __attribute__((ext_vector_type(8)))  __bf16 v8bf;
typedef __attribute__((ext_vector_type(8)))  float  v8f;

#define NEG_SLOPE 0.2f
#define BN_EPS    1e-5f

// ---------- order-preserving float <-> uint key (for atomicMax-based segment_max) ----------
__device__ __forceinline__ unsigned fkey(float f) {
  unsigned u = __float_as_uint(f);
  return (u >> 31) ? ~u : (u | 0x80000000u);
}
__device__ __forceinline__ float funkey(unsigned k) {
  return (k >> 31) ? __uint_as_float(k & 0x7fffffffu) : __uint_as_float(~k);
}

// ---------- small utility kernels ----------
__global__ void k_cvt_bf16(const float* __restrict__ src, __bf16* __restrict__ dst, int n) {
  int i = blockIdx.x * blockDim.x + threadIdx.x;
  if (i < n) dst[i] = (__bf16)src[i];
}

// W [K,C] f32 row-major  ->  Wt [C,K] bf16 row-major (so B fragments are contiguous in K)
__global__ void k_cvt_wT(const float* __restrict__ W, __bf16* __restrict__ Wt, int K, int C) {
  int i = blockIdx.x * blockDim.x + threadIdx.x;
  if (i >= K * C) return;
  int k = i / C, c = i % C;
  Wt[c * K + k] = (__bf16)W[i];
}

__global__ void k_fill_f32(float* __restrict__ p, float v, int n) {
  int i = blockIdx.x * blockDim.x + threadIdx.x;
  if (i < n) p[i] = v;
}
__global__ void k_fill_u32(unsigned* __restrict__ p, unsigned v, int n) {
  int i = blockIdx.x * blockDim.x + threadIdx.x;
  if (i < n) p[i] = v;
}
__global__ void k_fill_bias(float* __restrict__ p, const float* __restrict__ bias, int n, int C) {
  int i = blockIdx.x * blockDim.x + threadIdx.x;
  if (i < n) p[i] = bias[i % C];
}

// ---------- WMMA bf16 GEMM:  C[M, NT*16] = A[M,128] @ Bt[NT*16, 128]^T (+ bias) ----------
// One wave computes one 16-row tile across all NT*16 output columns.
template <int NT>
__global__ void k_gemm_bf16(const __bf16* __restrict__ A, const __bf16* __restrict__ Bt,
                            const float* __restrict__ bias, float* __restrict__ C, int M) {
  const int ncols = NT * 16;
  const int wave  = threadIdx.x >> 5;
  const int lane  = threadIdx.x & 31;
  const int tile  = blockIdx.x * (blockDim.x >> 5) + wave;
  const int m0    = tile * 16;
  if (m0 >= M) return;
  const int r  = lane & 15;   // row (A) / column (B) within tile
  const int hs = lane >> 4;   // lane half-select

  v8f acc[NT];
#pragma unroll
  for (int t = 0; t < NT; ++t) acc[t] = (v8f)(0.0f);

  const __bf16* arow = A + (size_t)(m0 + r) * 128;
#pragma unroll
  for (int kb = 0; kb < 4; ++kb) {
    // A 16x32 bf16 fragment: lanes 0-15 hold K 0..7 & 16..23, lanes 16-31 hold K 8..15 & 24..31
    const __bf16* ap = arow + kb * 32 + 8 * hs;
    v8bf alo = *(const v8bf*)(ap);
    v8bf ahi = *(const v8bf*)(ap + 16);
    v16bf a = __builtin_shufflevector(alo, ahi, 0,1,2,3,4,5,6,7,8,9,10,11,12,13,14,15);
#pragma unroll
    for (int t = 0; t < NT; ++t) {
      // B 32x16 bf16 fragment: lanes 0-15 hold K 0..7 & 16..23, lanes 16-31 hold K 8..15 & 24..31 of column t*16+r
      const __bf16* bp = Bt + (size_t)(t * 16 + r) * 128 + kb * 32 + 8 * hs;
      v8bf blo = *(const v8bf*)(bp);
      v8bf bhi = *(const v8bf*)(bp + 16);
      v16bf b = __builtin_shufflevector(blo, bhi, 0,1,2,3,4,5,6,7,8,9,10,11,12,13,14,15);
      acc[t] = __builtin_amdgcn_wmma_f32_16x16x32_bf16(false, a, false, b,
                                                       (short)0, acc[t], false, false);
    }
  }
  // C/D layout: VGPR j, lane l -> row = j + 8*(l>>4), col = l&15
#pragma unroll
  for (int t = 0; t < NT; ++t) {
    const int col = t * 16 + r;
    const float bv = bias ? bias[col] : 0.0f;
#pragma unroll
    for (int j = 0; j < 8; ++j) {
      int row = m0 + j + 8 * hs;
      C[(size_t)row * ncols + col] = acc[t][j] + bv;
    }
  }
}

// ---------- attention coefficients: a[n,h] = dot(xl[n,h*32:(h+1)*32], att[h]) ----------
__global__ void k_att(const float* __restrict__ xl, const float* __restrict__ attS,
                      const float* __restrict__ attD, float* __restrict__ aS,
                      float* __restrict__ aD, int N, int H) {
  int i = blockIdx.x * blockDim.x + threadIdx.x;   // i = n*H + h
  if (i >= N * H) return;
  int h = i % H;
  const float* row = xl + (size_t)i * 32;          // n*H*32 + h*32 == i*32
  const float* as = attS + h * 32;
  const float* ad = attD + h * 32;
  float s0 = 0.f, s1 = 0.f;
#pragma unroll
  for (int c = 0; c < 32; ++c) { float v = row[c]; s0 += v * as[c]; s1 += v * ad[c]; }
  aS[i] = s0;
  aD[i] = s1;
}

// ---------- edge pass 1: alpha = leaky_relu(a_src[src]+a_dst[dst]); segment max ----------
template <int H>
__global__ void k_edge_max(const int* __restrict__ eidx, int E, int Etot,
                           const float* __restrict__ aS, const float* __restrict__ aD,
                           float* __restrict__ buf, unsigned* __restrict__ akey) {
  int e = blockIdx.x * blockDim.x + threadIdx.x;
  if (e >= Etot) return;
  int s, d;
  if (e < E) { s = eidx[e]; d = eidx[E + e]; } else { s = d = e - E; }
#pragma unroll
  for (int i = 0; i < H; ++i) {
    float al = aS[(size_t)s * H + i] + aD[(size_t)d * H + i];
    al = (al > 0.f) ? al : NEG_SLOPE * al;
    buf[(size_t)e * H + i] = al;
    atomicMax(&akey[(size_t)d * H + i], fkey(al));
  }
}

// ---------- edge pass 2: ex = exp(alpha - amax[dst]); segment sum ----------
template <int H>
__global__ void k_edge_exp(const int* __restrict__ eidx, int E, int Etot,
                           float* __restrict__ buf, const unsigned* __restrict__ akey,
                           float* __restrict__ den) {
  int e = blockIdx.x * blockDim.x + threadIdx.x;
  if (e >= Etot) return;
  int d = (e < E) ? eidx[E + e] : (e - E);
#pragma unroll
  for (int i = 0; i < H; ++i) {
    float amax = funkey(akey[(size_t)d * H + i]);
    float ex = expf(buf[(size_t)e * H + i] - amax);
    buf[(size_t)e * H + i] = ex;
    atomicAdd(&den[(size_t)d * H + i], ex);
  }
}

// ---------- edge pass 3: out[dst] += (ex/denom) * xl[src]   (one wave per edge) ----------
template <int H>
__global__ void k_edge_agg(const int* __restrict__ eidx, int E, int Etot,
                           const float* __restrict__ buf, const float* __restrict__ den,
                           const float* __restrict__ xl, float* __restrict__ out) {
  const int lane = threadIdx.x & 31;
  int e = blockIdx.x * (blockDim.x >> 5) + (threadIdx.x >> 5);
  if (e >= Etot) return;
  int s, d;
  if (e < E) { s = eidx[e]; d = eidx[E + e]; } else { s = d = e - E; }
#pragma unroll
  for (int i = 0; i < H; ++i) {
    float w = buf[(size_t)e * H + i] / (den[(size_t)d * H + i] + 1e-16f);
    float v = xl[(size_t)s * (H * 32) + i * 32 + lane] * w;
    atomicAdd(&out[(size_t)d * (H * 32) + i * 32 + lane], v);
  }
}

// ---------- batchnorm statistics: per-channel sum and sum of squares ----------
__global__ void k_bn_stats(const float* __restrict__ X, float* __restrict__ S,
                           float* __restrict__ Q, int N, int C) {
  int rpb = blockDim.x / C;
  int j = threadIdx.x % C;
  int r0 = blockIdx.x * rpb + threadIdx.x / C;
  int stride = gridDim.x * rpb;
  float s = 0.f, q = 0.f;
  for (int r = r0; r < N; r += stride) { float v = X[(size_t)r * C + j]; s += v; q += v * v; }
  atomicAdd(&S[j], s);
  atomicAdd(&Q[j], q);
}

// ---------- BN -> ReLU -> +residual -> bf16 (feeds layer-1 GEMM) ----------
__global__ void k_bn_relu_res_cvt(const float* __restrict__ X, const float* __restrict__ S,
                                  const float* __restrict__ Q, const float* __restrict__ g,
                                  const float* __restrict__ bt, const float* __restrict__ res,
                                  __bf16* __restrict__ hb, int N, int C) {
  int i = blockIdx.x * blockDim.x + threadIdx.x;
  if (i >= N * C) return;
  int j = i % C;
  float inv = 1.0f / (float)N;
  float mu  = S[j] * inv;
  float var = Q[j] * inv - mu * mu;
  float v = g[j] * (X[i] - mu) * rsqrtf(var + BN_EPS) + bt[j];
  v = (v > 0.f) ? v : 0.f;
  hb[i] = (__bf16)(v + res[i]);
}

// ---------- BN -> ReLU -> final output ----------
__global__ void k_bn_relu_out(const float* __restrict__ X, const float* __restrict__ S,
                              const float* __restrict__ Q, const float* __restrict__ g,
                              const float* __restrict__ bt, float* __restrict__ out, int N, int C) {
  int i = blockIdx.x * blockDim.x + threadIdx.x;
  if (i >= N * C) return;
  int j = i % C;
  float inv = 1.0f / (float)N;
  float mu  = S[j] * inv;
  float var = Q[j] * inv - mu * mu;
  float v = g[j] * (X[i] - mu) * rsqrtf(var + BN_EPS) + bt[j];
  out[i] = (v > 0.f) ? v : 0.f;
}

static inline int cdiv(int a, int b) { return (a + b - 1) / b; }

extern "C" void kernel_launch(void* const* d_in, const int* in_sizes, int n_in,
                              void* d_out, int out_size, void* d_ws, size_t ws_size,
                              hipStream_t stream) {
  (void)n_in; (void)out_size; (void)ws_size;
  const float* x     = (const float*)d_in[0];
  const int*   eidx  = (const int*)d_in[1];
  const float* W1    = (const float*)d_in[2];
  const float* attS1 = (const float*)d_in[3];
  const float* attD1 = (const float*)d_in[4];
  const float* b1    = (const float*)d_in[5];
  const float* g1    = (const float*)d_in[6];
  const float* bt1   = (const float*)d_in[7];
  const float* Wr0   = (const float*)d_in[8];
  const float* br0   = (const float*)d_in[9];
  const float* W2    = (const float*)d_in[10];
  const float* attS2 = (const float*)d_in[11];
  const float* attD2 = (const float*)d_in[12];
  const float* b2    = (const float*)d_in[13];
  const float* g2    = (const float*)d_in[14];
  const float* bt2   = (const float*)d_in[15];

  const int N    = in_sizes[0] / 128;
  const int E    = in_sizes[1] / 2;
  const int Etot = E + N;

  // ---- workspace layout (256B aligned), with aliasing of dead buffers ----
  char* ws = (char*)d_ws;
  size_t off = 0;
  auto alloc = [&](size_t bytes) -> char* {
    size_t start = (off + 255) & ~(size_t)255;
    off = start + bytes;
    return ws + start;
  };
  __bf16* xb    = (__bf16*)alloc((size_t)N * 128 * 2);   // x in bf16; reused as hb later
  __bf16* w1t   = (__bf16*)alloc(128 * 128 * 2);
  __bf16* wr0t  = (__bf16*)alloc(128 * 128 * 2);
  __bf16* w2t   = (__bf16*)alloc(32 * 128 * 2);
  float*  xl    = (float*)alloc((size_t)N * 128 * 4);    // reused as xl2 later
  float*  ident = (float*)alloc((size_t)N * 128 * 4);    // reused as out1 later
  float*  out0  = (float*)alloc((size_t)N * 128 * 4);
  float*  aS1   = (float*)alloc((size_t)N * 4 * 4);      // reused as aS2
  float*  aD1   = (float*)alloc((size_t)N * 4 * 4);      // reused as aD2
  unsigned* akey1 = (unsigned*)alloc((size_t)N * 4 * 4); // reused as akey2
  float*  den1  = (float*)alloc((size_t)N * 4 * 4);      // reused as den2
  float*  ebuf  = (float*)alloc((size_t)Etot * 4 * 4);   // alpha/ex buffer, reused for layer 1
  float*  SQ    = (float*)alloc(320 * 4);                // S1[128] Q1[128] S2[32] Q2[32]
  float* S1 = SQ, *Q1 = SQ + 128, *S2 = SQ + 256, *Q2 = SQ + 288;
  // aliases
  __bf16* hb   = xb;     // layer-0 output in bf16 (x dead after GEMMs)
  float*  xl2  = xl;     // layer-1 transformed features (xl dead after layer-0 agg)
  float*  out1 = ident;  // layer-1 aggregation (identity dead after bn_relu_res)
  float*  aS2  = aS1, *aD2 = aD1, *den2 = den1;
  unsigned* akey2 = akey1;
  float*  ebuf2 = ebuf;

  const int B = 256;

  // ---- precision conversion / weight transpose ----
  k_cvt_bf16<<<cdiv(N * 128, B), B, 0, stream>>>(x, xb, N * 128);
  k_cvt_wT  <<<cdiv(128 * 128, B), B, 0, stream>>>(W1, w1t, 128, 128);
  k_cvt_wT  <<<cdiv(128 * 128, B), B, 0, stream>>>(Wr0, wr0t, 128, 128);
  k_cvt_wT  <<<cdiv(128 * 32, B), B, 0, stream>>>(W2, w2t, 128, 32);

  // ---- init layer-0 accumulators ----
  k_fill_bias<<<cdiv(N * 128, B), B, 0, stream>>>(out0, b1, N * 128, 128);
  k_fill_u32 <<<cdiv(N * 4, B), B, 0, stream>>>(akey1, 0u, N * 4);
  k_fill_f32 <<<cdiv(N * 4, B), B, 0, stream>>>(den1, 0.f, N * 4);
  k_fill_f32 <<<1, 320, 0, stream>>>(SQ, 0.f, 320);

  // ---- layer-0 GEMMs (WMMA bf16) ----
  {
    int tiles = cdiv(N, 16);
    int grid = cdiv(tiles, 4);  // 4 waves / block
    k_gemm_bf16<8><<<grid, 128, 0, stream>>>(xb, w1t, nullptr, xl, N);
    k_gemm_bf16<8><<<grid, 128, 0, stream>>>(xb, wr0t, br0, ident, N);
  }

  // ---- layer-0 attention + edge softmax + aggregation ----
  k_att<<<cdiv(N * 4, B), B, 0, stream>>>(xl, attS1, attD1, aS1, aD1, N, 4);
  k_edge_max<4><<<cdiv(Etot, B), B, 0, stream>>>(eidx, E, Etot, aS1, aD1, ebuf, akey1);
  k_edge_exp<4><<<cdiv(Etot, B), B, 0, stream>>>(eidx, E, Etot, ebuf, akey1, den1);
  k_edge_agg<4><<<cdiv(Etot, 8), 256, 0, stream>>>(eidx, E, Etot, ebuf, den1, xl, out0);

  // ---- BN -> ReLU -> residual -> bf16 h ----
  k_bn_stats<<<1024, 128, 0, stream>>>(out0, S1, Q1, N, 128);
  k_bn_relu_res_cvt<<<cdiv(N * 128, B), B, 0, stream>>>(out0, S1, Q1, g1, bt1, ident, hb, N, 128);

  // ---- init layer-1 accumulators (identity is dead now; safe to alias out1) ----
  k_fill_bias<<<cdiv(N * 32, B), B, 0, stream>>>(out1, b2, N * 32, 32);
  k_fill_u32 <<<cdiv(N, B), B, 0, stream>>>(akey2, 0u, N);
  k_fill_f32 <<<cdiv(N, B), B, 0, stream>>>(den2, 0.f, N);

  // ---- layer-1 GEMM (WMMA bf16) ----
  {
    int tiles = cdiv(N, 16);
    int grid = cdiv(tiles, 4);
    k_gemm_bf16<2><<<grid, 128, 0, stream>>>(hb, w2t, nullptr, xl2, N);
  }

  // ---- layer-1 attention + edge softmax + aggregation ----
  k_att<<<cdiv(N, B), B, 0, stream>>>(xl2, attS2, attD2, aS2, aD2, N, 1);
  k_edge_max<1><<<cdiv(Etot, B), B, 0, stream>>>(eidx, E, Etot, aS2, aD2, ebuf2, akey2);
  k_edge_exp<1><<<cdiv(Etot, B), B, 0, stream>>>(eidx, E, Etot, ebuf2, akey2, den2);
  k_edge_agg<1><<<cdiv(Etot, 8), 256, 0, stream>>>(eidx, E, Etot, ebuf2, den2, xl2, out1);

  // ---- final BN -> ReLU -> d_out ----
  k_bn_stats<<<1024, 128, 0, stream>>>(out1, S2, Q2, N, 32);
  k_bn_relu_out<<<cdiv(N * 32, B), B, 0, stream>>>(out1, S2, Q2, g2, bt2, (float*)d_out, N, 32);
}
